// MultiheadAttention_13975823581383
// MI455X (gfx1250) — compile-verified, matrix operand order checked by reading, not run on hardware
//
#include <hip/hip_runtime.h>
#include <hip/hip_bf16.h>
#include <math.h>

#define EMB   1024
#define HEADS 16
#define HD    64
#define BATCH 2
#define SEQ   2048

typedef __attribute__((ext_vector_type(16))) __bf16 v16bf;
typedef __attribute__((ext_vector_type(8)))  __bf16 v8bf;
typedef __attribute__((ext_vector_type(8)))  float  v8f;

__device__ __forceinline__ v8f wmma_bf16(v16bf a, v16bf b, v8f c) {
  // D = A(16x32 bf16) * B(32x16 bf16) + C(16x16 f32)
  return __builtin_amdgcn_wmma_f32_16x16x32_bf16(
      /*neg_a=*/false, a, /*neg_b=*/false, b,
      /*c_mod=*/(short)0, c, /*reuse_a=*/false, /*reuse_b=*/false);
}

// A/B operand pack: lane holds row(M)/col(N) = lane%16; with half = lane/16,
// element e in 0..7 -> K = half*8 + e ; e in 8..15 -> K = 16 + half*8 + (e-8).
// `p` points at (row base + half*8) so the two runs are p[0..7] and p[16..23].
__device__ __forceinline__ v16bf load16_bf16(const __bf16* p) {
  v16bf a;
#pragma unroll
  for (int i = 0; i < 8; ++i) { a[i] = p[i]; a[i + 8] = p[16 + i]; }
  return a;
}

// ---------------------------------------------------------------------------
// Kernel 0: one-shot f32 -> bf16 conversion (bandwidth bound, no redundancy).
// n must be a multiple of 2048 (all tensors here are).
// ---------------------------------------------------------------------------
__global__ __launch_bounds__(256) void cvt_f32_bf16_kernel(
    const float* __restrict__ src, __bf16* __restrict__ dst) {
  const size_t i = ((size_t)blockIdx.x * 256 + threadIdx.x) * 8;
  v8f a = *(const v8f*)(src + i);
  v8bf o;
#pragma unroll
  for (int j = 0; j < 8; ++j) o[j] = (__bf16)a[j];
  *(v8bf*)(dst + i) = o;
}

// ---------------------------------------------------------------------------
// Kernel 1: Q/K/V projections, all-bf16 operands. Y = X @ W^T + b.
// z==0 -> Qh [B,S,D], z==1 -> Kh [B,S,D], z==2 -> Vt transposed [B,H,HD,S].
// Each wave: 16x32 output (2 col tiles), 4 independent accumulators -> A
// operands reused across both col tiles, no WMMA->WMMA hazard chains.
// ---------------------------------------------------------------------------
__global__ __launch_bounds__(256) void proj_qkv_kernel(
    const __bf16* __restrict__ Xq, const __bf16* __restrict__ Xk, const __bf16* __restrict__ Xv,
    const __bf16* __restrict__ Wqb, const __bf16* __restrict__ Wkb, const __bf16* __restrict__ Wvb,
    const float* __restrict__ bq, const float* __restrict__ bk, const float* __restrict__ bv,
    __bf16* __restrict__ Qh, __bf16* __restrict__ Kh, __bf16* __restrict__ Vt) {
  const int lane = threadIdx.x & 31, wave = threadIdx.x >> 5;
  const int l16 = lane & 15, half = lane >> 4;
  const int cg = blockIdx.x * 8 + wave;  // col group 0..31 (32 cols each)
  const int rt = blockIdx.y;             // row tile 0..255
  const int z  = blockIdx.z;

  const __bf16* X  = (z == 0) ? Xq  : (z == 1) ? Xk  : Xv;
  const __bf16* W  = (z == 0) ? Wqb : (z == 1) ? Wkb : Wvb;
  const float*  bb = (z == 0) ? bq  : (z == 1) ? bk  : bv;

  const int m0 = rt * 16, n0 = cg * 32;
  const __bf16* xrow = X + (size_t)(m0 + l16) * EMB + half * 8;       // A rows
  const __bf16* w0   = W + (size_t)(n0 + l16) * EMB + half * 8;       // B col tile 0
  const __bf16* w1   = W + (size_t)(n0 + 16 + l16) * EMB + half * 8;  // B col tile 1

  v8f c0a = {}, c0b = {}, c1a = {}, c1b = {};
#pragma unroll 2
  for (int kk = 0; kk < EMB; kk += 64) {
    v16bf a0 = load16_bf16(xrow + kk);
    v16bf a1 = load16_bf16(xrow + kk + 32);
    c0a = wmma_bf16(a0, load16_bf16(w0 + kk), c0a);
    c1a = wmma_bf16(a0, load16_bf16(w1 + kk), c1a);
    c0b = wmma_bf16(a1, load16_bf16(w0 + kk + 32), c0b);
    c1b = wmma_bf16(a1, load16_bf16(w1 + kk + 32), c1b);
  }
  const v8f c0 = c0a + c0b;
  const v8f c1 = c1a + c1b;
  const float bias0 = bb[n0 + l16];
  const float bias1 = bb[n0 + 16 + l16];

  if (z < 2) {
    __bf16* Y = (z == 0) ? Qh : Kh;
#pragma unroll
    for (int r = 0; r < 8; ++r) {
      const int m = m0 + half * 8 + r;  // C layout: VGPR r -> M = r + 8*half
      Y[(size_t)m * EMB + n0 + l16]      = (__bf16)(c0[r] + bias0);
      Y[(size_t)m * EMB + n0 + 16 + l16] = (__bf16)(c1[r] + bias1);
    }
  } else {
#pragma unroll
    for (int t = 0; t < 2; ++t) {
      const int n = n0 + t * 16 + l16, h = n >> 6, d = n & 63;
      const v8f cc = t ? c1 : c0;
      const float bias = t ? bias1 : bias0;
#pragma unroll
      for (int r = 0; r < 8; ++r) {
        const int m = m0 + half * 8 + r;
        const int bidx = m >> 11, s = m & (SEQ - 1);
        Vt[((size_t)(bidx * HEADS + h) * HD + d) * SEQ + s] = (__bf16)(cc[r] + bias);
      }
    }
  }
}

// ---------------------------------------------------------------------------
// Kernel 2: fused causal attention. One wave per 16-row q tile.
// Sweep 1: online row max/sum over QK^T/8 (no storage).
// Sweep 2: recompute scores, write normalized attn (f32, zeros past diag),
// stage P via per-wave LDS to A-operand layout, accumulate O = P @ V (WMMA).
// ---------------------------------------------------------------------------
__global__ __launch_bounds__(256) void attn_kernel(
    const __bf16* __restrict__ Qh, const __bf16* __restrict__ Kh,
    const __bf16* __restrict__ Vt, float* __restrict__ attn,
    __bf16* __restrict__ AttnOut) {
  __shared__ __bf16 psh[8][16 * 32];  // per-wave 16x32 P staging (row-major)

  const int lane = threadIdx.x & 31, wave = threadIdx.x >> 5;
  const int l16 = lane & 15, half = lane >> 4;
  const int qt = blockIdx.x * 8 + wave;  // q tile 0..127
  const int bh = blockIdx.y;             // 0..31
  const int b = bh >> 4, h = bh & 15;
  const int q0 = qt * 16;
  const float scale = 0.125f;  // 1/sqrt(hd)

  // Resident Q operand (16 rows x hd=64 -> two K=32 chunks)
  const __bf16* qrow = Qh + ((size_t)(b * SEQ + q0 + l16)) * EMB + h * HD + half * 8;
  const v16bf qa0 = load16_bf16(qrow);
  const v16bf qa1 = load16_bf16(qrow + 32);

  float runmax[8], runsum[8];
#pragma unroll
  for (int r = 0; r < 8; ++r) { runmax[r] = -__builtin_inff(); runsum[r] = 0.f; }

  // ---- Sweep 1: online max/sum ----
  for (int kt = 0; kt <= qt; ++kt) {
    const __bf16* krow = Kh + ((size_t)(b * SEQ + kt * 16 + l16)) * EMB + h * HD + half * 8;
    v8f sa = {}, sb = {};
    sa = wmma_bf16(qa0, load16_bf16(krow), sa);       // independent pair:
    sb = wmma_bf16(qa1, load16_bf16(krow + 32), sb);  // no WMMA->WMMA hazard
    const bool diag = (kt == qt);
#pragma unroll
    for (int r = 0; r < 8; ++r) {
      float s = (sa[r] + sb[r]) * scale;
      if (diag && (l16 > half * 8 + r)) s = -__builtin_inff();
      float tm = s;
#pragma unroll
      for (int mm = 1; mm < 16; mm <<= 1) tm = fmaxf(tm, __shfl_xor(tm, mm, 32));
      const float nm = fmaxf(runmax[r], tm);
      const float corr = __expf(runmax[r] - nm);
      float ts = __expf(s - nm);
#pragma unroll
      for (int mm = 1; mm < 16; mm <<= 1) ts += __shfl_xor(ts, mm, 32);
      runsum[r] = runsum[r] * corr + ts;
      runmax[r] = nm;
    }
  }
  float rinv[8];
#pragma unroll
  for (int r = 0; r < 8; ++r) rinv[r] = 1.0f / runsum[r];

  // ---- Sweep 2: recompute, emit attn, accumulate O ----
  v8f o0 = {}, o1 = {}, o2 = {}, o3 = {};
  float* arow = attn + (size_t)bh * SEQ * SEQ;

  for (int c = 0; c < SEQ / 32; ++c) {
    const int kb = c * 32;
#pragma unroll
    for (int t = 0; t < 2; ++t) {
      const int kt = 2 * c + t;
      float p[8];
      if (kt <= qt) {
        const __bf16* krow = Kh + ((size_t)(b * SEQ + kt * 16 + l16)) * EMB + h * HD + half * 8;
        v8f sa = {}, sb = {};
        sa = wmma_bf16(qa0, load16_bf16(krow), sa);
        sb = wmma_bf16(qa1, load16_bf16(krow + 32), sb);
        const bool diag = (kt == qt);
#pragma unroll
        for (int r = 0; r < 8; ++r) {
          const bool masked = diag && (l16 > half * 8 + r);
          p[r] = masked ? 0.f
                        : __expf((sa[r] + sb[r]) * scale - runmax[r]) * rinv[r];
        }
      } else {
#pragma unroll
        for (int r = 0; r < 8; ++r) p[r] = 0.f;
      }
#pragma unroll
      for (int r = 0; r < 8; ++r) {
        arow[(size_t)(q0 + half * 8 + r) * SEQ + kt * 16 + l16] = p[r];
        psh[wave][(half * 8 + r) * 32 + t * 16 + l16] = (__bf16)p[r];
      }
    }
    // DS ops are in-order within a wave; wait + compiler fence before re-read.
    asm volatile("s_wait_dscnt 0x0" ::: "memory");
    v16bf pa;
#pragma unroll
    for (int i = 0; i < 8; ++i) {
      pa[i]     = psh[wave][l16 * 32 + half * 8 + i];
      pa[i + 8] = psh[wave][l16 * 32 + 16 + half * 8 + i];
    }
    if (2 * c <= qt) {  // wave-uniform: P is nonzero somewhere in this chunk
      const __bf16* vbase = Vt + (size_t)bh * HD * SEQ + kb + half * 8;
      o0 = wmma_bf16(pa, load16_bf16(vbase + (size_t)(0 * 16 + l16) * SEQ), o0);
      o1 = wmma_bf16(pa, load16_bf16(vbase + (size_t)(1 * 16 + l16) * SEQ), o1);
      o2 = wmma_bf16(pa, load16_bf16(vbase + (size_t)(2 * 16 + l16) * SEQ), o2);
      o3 = wmma_bf16(pa, load16_bf16(vbase + (size_t)(3 * 16 + l16) * SEQ), o3);
    }
  }

  __bf16* orow = AttnOut + (size_t)(b * SEQ + q0) * EMB + h * HD;
#pragma unroll
  for (int dblk = 0; dblk < 4; ++dblk) {
    v8f o = (dblk == 0) ? o0 : (dblk == 1) ? o1 : (dblk == 2) ? o2 : o3;
#pragma unroll
    for (int r = 0; r < 8; ++r)
      orow[(size_t)(half * 8 + r) * EMB + dblk * 16 + l16] = (__bf16)o[r];
  }
}

// ---------------------------------------------------------------------------
// Kernel 3: output projection, all-bf16 operands. out = AttnOut @ Wo^T + bo.
// ---------------------------------------------------------------------------
__global__ __launch_bounds__(256) void proj_out_kernel(
    const __bf16* __restrict__ Xin, const __bf16* __restrict__ Wob,
    const float* __restrict__ bo, float* __restrict__ Y) {
  const int lane = threadIdx.x & 31, wave = threadIdx.x >> 5;
  const int l16 = lane & 15, half = lane >> 4;
  const int cg = blockIdx.x * 8 + wave;
  const int rt = blockIdx.y;
  const int m0 = rt * 16, n0 = cg * 32;

  const __bf16* xrow = Xin + (size_t)(m0 + l16) * EMB + half * 8;
  const __bf16* w0   = Wob + (size_t)(n0 + l16) * EMB + half * 8;
  const __bf16* w1   = Wob + (size_t)(n0 + 16 + l16) * EMB + half * 8;

  v8f c0a = {}, c0b = {}, c1a = {}, c1b = {};
#pragma unroll 2
  for (int kk = 0; kk < EMB; kk += 64) {
    v16bf a0 = load16_bf16(xrow + kk);
    v16bf a1 = load16_bf16(xrow + kk + 32);
    c0a = wmma_bf16(a0, load16_bf16(w0 + kk), c0a);
    c1a = wmma_bf16(a0, load16_bf16(w1 + kk), c1a);
    c0b = wmma_bf16(a1, load16_bf16(w0 + kk + 32), c0b);
    c1b = wmma_bf16(a1, load16_bf16(w1 + kk + 32), c1b);
  }
  const v8f c0 = c0a + c0b;
  const v8f c1 = c1a + c1b;
  const float bias0 = bo[n0 + l16];
  const float bias1 = bo[n0 + 16 + l16];
#pragma unroll
  for (int r = 0; r < 8; ++r) {
    const int m = m0 + half * 8 + r;
    Y[(size_t)m * EMB + n0 + l16]      = c0[r] + bias0;
    Y[(size_t)m * EMB + n0 + 16 + l16] = c1[r] + bias1;
  }
}

extern "C" void kernel_launch(void* const* d_in, const int* in_sizes, int n_in,
                              void* d_out, int out_size, void* d_ws, size_t ws_size,
                              hipStream_t stream) {
  const float* q  = (const float*)d_in[0];
  const float* k  = (const float*)d_in[1];
  const float* v  = (const float*)d_in[2];
  const float* Wq = (const float*)d_in[3];
  const float* bq = (const float*)d_in[4];
  const float* Wk = (const float*)d_in[5];
  const float* bk = (const float*)d_in[6];
  const float* Wv = (const float*)d_in[7];
  const float* bv = (const float*)d_in[8];
  const float* Wo = (const float*)d_in[9];
  const float* bo = (const float*)d_in[10];

  float* out  = (float*)d_out;                          // [B,S,D]
  float* attn = out + (size_t)BATCH * SEQ * EMB;        // [B*H,S,S]

  const size_t tsz = (size_t)BATCH * SEQ * EMB;         // 4M elems
  const size_t wsz = (size_t)EMB * EMB;                 // 1M elems
  __bf16* Qh      = (__bf16*)d_ws;                      // [B,S,D]
  __bf16* Kh      = Qh + tsz;                           // [B,S,D]
  __bf16* Vt      = Kh + tsz;                           // [B,H,HD,S]
  __bf16* AttnOut = Vt + tsz;                           // [B,S,D]
  __bf16* Xq      = AttnOut + tsz;                      // bf16 copies of inputs
  __bf16* Xk      = Xq + tsz;
  __bf16* Xv      = Xk + tsz;
  __bf16* Wqb     = Xv + tsz;                           // bf16 copies of weights
  __bf16* Wkb     = Wqb + wsz;
  __bf16* Wvb     = Wkb + wsz;
  __bf16* Wob     = Wvb + wsz;

  dim3 blk(256, 1, 1);

  // One-shot f32->bf16 conversions (each n is a multiple of 2048).
  hipLaunchKernelGGL(cvt_f32_bf16_kernel, dim3(tsz / 2048), blk, 0, stream, q, Xq);
  hipLaunchKernelGGL(cvt_f32_bf16_kernel, dim3(tsz / 2048), blk, 0, stream, k, Xk);
  hipLaunchKernelGGL(cvt_f32_bf16_kernel, dim3(tsz / 2048), blk, 0, stream, v, Xv);
  hipLaunchKernelGGL(cvt_f32_bf16_kernel, dim3(wsz / 2048), blk, 0, stream, Wq, Wqb);
  hipLaunchKernelGGL(cvt_f32_bf16_kernel, dim3(wsz / 2048), blk, 0, stream, Wk, Wkb);
  hipLaunchKernelGGL(cvt_f32_bf16_kernel, dim3(wsz / 2048), blk, 0, stream, Wv, Wvb);
  hipLaunchKernelGGL(cvt_f32_bf16_kernel, dim3(wsz / 2048), blk, 0, stream, Wo, Wob);

  dim3 g1(4, (BATCH * SEQ) / 16, 3);  // 8 waves/block * 4 = 32 col groups
  hipLaunchKernelGGL(proj_qkv_kernel, g1, blk, 0, stream,
                     Xq, Xk, Xv, Wqb, Wkb, Wvb, bq, bk, bv, Qh, Kh, Vt);

  dim3 g2((SEQ / 16) / 8, BATCH * HEADS, 1);
  hipLaunchKernelGGL(attn_kernel, g2, blk, 0, stream, Qh, Kh, Vt, attn, AttnOut);

  dim3 g3(4, (BATCH * SEQ) / 16, 1);
  hipLaunchKernelGGL(proj_out_kernel, g3, blk, 0, stream, AttnOut, Wob, bo, out);
}
